// GCNNet_5634997092668
// MI455X (gfx1250) — compile-verified, matrix operand order checked by reading, not run on hardware
//
#include <hip/hip_runtime.h>
#include <stdint.h>

typedef __attribute__((ext_vector_type(16))) __bf16 v16bf;
typedef __attribute__((ext_vector_type(8)))  __bf16 v8bf;
typedef __attribute__((ext_vector_type(8)))  float  v8f;

#define NEG_SLOPE 0.01f

// ---------------------------------------------------------------------------
// WMMA GEMM: C[M,N] = A[M,K] @ B[K,N]
//   A:  bf16 row-major [M,K]
//   Bt: bf16 row-major [N,K]  (B transposed, so fragment loads mirror A's)
//   C:  f32  row-major [M,N]
// One wave computes one 16x16 tile, K-loop in steps of 32 (v_wmma_f32_16x16x32_bf16).
// Fragment layout per CDNA5 ISA 7.12.2 (16-bit A 16x32):
//   lanes 0-15  : row M=lane,    K = {k0+0..7,  k0+16..23}
//   lanes 16-31 : row M=lane-16, K = {k0+8..15, k0+24..31}
// Each half is 8 contiguous bf16 = one 16-byte load.
// ---------------------------------------------------------------------------
__global__ __launch_bounds__(256) void gcn_wmma_gemm_bf16(
    const __bf16* __restrict__ A, const __bf16* __restrict__ Bt,
    float* __restrict__ C, int M, int N, int K)
{
    const int lane  = threadIdx.x & 31;
    const int wave  = threadIdx.x >> 5;
    const int tilesN = N >> 4;
    const int tile  = blockIdx.x * 8 + wave;
    const int totalTiles = (M >> 4) * tilesN;
    if (tile >= totalTiles) return;

    const int tm   = tile / tilesN;
    const int tn   = tile % tilesN;
    const int half = lane >> 4;          // 0 or 1
    const int l15  = lane & 15;
    const int kOff = half * 8;

    const __bf16* arow = A  + (size_t)(tm * 16 + l15) * K;
    const __bf16* brow = Bt + (size_t)(tn * 16 + l15) * K;

    v8f acc = {};
    for (int k0 = 0; k0 < K; k0 += 32) {
        v8bf alo = *(const v8bf*)(arow + k0 + kOff);
        v8bf ahi = *(const v8bf*)(arow + k0 + 16 + kOff);
        v8bf blo = *(const v8bf*)(brow + k0 + kOff);
        v8bf bhi = *(const v8bf*)(brow + k0 + 16 + kOff);
        v16bf a, b;
#pragma unroll
        for (int i = 0; i < 8; ++i) {
            a[i] = alo[i]; a[i + 8] = ahi[i];
            b[i] = blo[i]; b[i + 8] = bhi[i];
        }
        acc = __builtin_amdgcn_wmma_f32_16x16x32_bf16(
            /*neg_a=*/false, a, /*neg_b=*/false, b,
            /*c_mod=*/(short)0, acc, /*reuse_a=*/false, /*reuse_b=*/false);
    }

    // C fragment: VGPR r -> M = tm*16 + r + half*8, N = tn*16 + (lane&15)
    float* cbase = C + (size_t)(tm * 16 + half * 8) * N + tn * 16 + l15;
#pragma unroll
    for (int r = 0; r < 8; ++r)
        cbase[(size_t)r * N] = acc[r];
}

// ---------------------------------------------------------------------------
// Degree / normalization helpers
// ---------------------------------------------------------------------------
__global__ void k_fill_ones(float* __restrict__ d, int n) {
    int i = blockIdx.x * blockDim.x + threadIdx.x;
    if (i < n) d[i] = 1.0f;                       // self-loop contribution
}

__global__ void k_deg_accum(const long long* __restrict__ dst,
                            float* __restrict__ deg, int e) {
    int i = blockIdx.x * blockDim.x + threadIdx.x;
    if (i < e) atomicAdd(&deg[dst[i]], 1.0f);
}

__global__ void k_rsqrt_inplace(float* __restrict__ d, int n) {
    int i = blockIdx.x * blockDim.x + threadIdx.x;
    if (i < n) {
        float v = d[i];
        d[i] = (v > 0.0f) ? rsqrtf(v) : 0.0f;
    }
}

// ---------------------------------------------------------------------------
// Weight transpose + f32 -> bf16:  W[K,N] -> Wt[N,K]
// ---------------------------------------------------------------------------
__global__ void k_transpose_cvt(const float* __restrict__ W,
                                __bf16* __restrict__ Wt, int K, int N) {
    int i = blockIdx.x * blockDim.x + threadIdx.x;
    if (i < K * N) {
        int k = i / N, n = i % N;
        Wt[(size_t)n * K + k] = (__bf16)W[i];
    }
}

// f32 -> bf16 (GEMM-1 input)
__global__ void k_cvt_bf16(const float* __restrict__ in,
                           __bf16* __restrict__ out, long long n) {
    long long i = (long long)blockIdx.x * blockDim.x + threadIdx.x;
    if (i < n) out[i] = (__bf16)in[i];
}

// fused leaky-relu + f32 -> bf16 (feeds the next GEMM)
__global__ void k_lrelu_cvt_bf16(const float* __restrict__ in,
                                 __bf16* __restrict__ out, long long n) {
    long long i = (long long)blockIdx.x * blockDim.x + threadIdx.x;
    if (i < n) {
        float v = in[i];
        out[i] = (__bf16)(v > 0.0f ? v : NEG_SLOPE * v);
    }
}

// ---------------------------------------------------------------------------
// Aggregation: out[i,f] = h[i,f]*dis[i]^2 + b[f]   (self-loop + bias init)
// then edge scatter adds h[src]*dis[src]*dis[dst] into out[dst].
// ---------------------------------------------------------------------------
__global__ void k_agg_init(const float* __restrict__ h,
                           const float* __restrict__ dis,
                           const float* __restrict__ bias,
                           float* __restrict__ out, int n, int f) {
    long long idx = (long long)blockIdx.x * blockDim.x + threadIdx.x;
    if (idx < (long long)n * f) {
        int i = (int)(idx / f), c = (int)(idx % f);
        float d = dis[i];
        out[idx] = h[idx] * d * d + bias[c];
    }
}

__global__ void k_edge_scatter(const float* __restrict__ h,
                               const float* __restrict__ dis,
                               const long long* __restrict__ src,
                               const long long* __restrict__ dst,
                               float* __restrict__ out, int e, int f) {
    const int groups = f >> 2;                      // float4 per thread
    long long idx = (long long)blockIdx.x * blockDim.x + threadIdx.x;
    if (idx >= (long long)e * groups) return;
    int ei = (int)(idx / groups);
    int g  = (int)(idx % groups) * 4;
    long long s = src[ei], d = dst[ei];
    float norm = dis[s] * dis[d];
    const float4 hv = *(const float4*)(h + s * (long long)f + g);
    float* o = out + d * (long long)f + g;
    atomicAdd(o + 0, hv.x * norm);
    atomicAdd(o + 1, hv.y * norm);
    atomicAdd(o + 2, hv.z * norm);
    atomicAdd(o + 3, hv.w * norm);
}

// ---------------------------------------------------------------------------
// Readout: first node of each graph (batch sorted, ids 0..G-1)
// ---------------------------------------------------------------------------
__global__ void k_gather_first(const float* __restrict__ h,
                               const long long* __restrict__ batch,
                               float* __restrict__ out, int n, int f) {
    int i = blockIdx.x * blockDim.x + threadIdx.x;
    if (i >= n) return;
    bool first = (i == 0) || (batch[i] != batch[i - 1]);
    if (first) {
        long long g = batch[i];
        for (int c = 0; c < f; ++c)
            out[g * f + c] = h[(long long)i * f + c];
    }
}

// ---------------------------------------------------------------------------
// Host-side orchestration
// ---------------------------------------------------------------------------
static inline unsigned blocks_for(long long n, int bs) {
    return (unsigned)((n + bs - 1) / bs);
}

extern "C" void kernel_launch(void* const* d_in, const int* in_sizes, int n_in,
                              void* d_out, int out_size, void* d_ws, size_t ws_size,
                              hipStream_t stream) {
    const int IN = 128, H = 256, OUT = 16;
    const int N = in_sizes[0] / IN;
    const int E = in_sizes[1] / 2;

    const float*     x  = (const float*)d_in[0];
    const long long* ei = (const long long*)d_in[1];
    const long long* batch = (const long long*)d_in[2];
    const float* W1 = (const float*)d_in[3];
    const float* b1 = (const float*)d_in[4];
    const float* W2 = (const float*)d_in[5];
    const float* b2 = (const float*)d_in[6];
    const float* W3 = (const float*)d_in[7];
    const float* b3 = (const float*)d_in[8];
    float* out = (float*)d_out;

    const long long* src = ei;
    const long long* dst = ei + E;

    // ---- workspace carve-out (256B aligned) ----
    char* ws = (char*)d_ws;
    size_t off = 0;
    auto carve = [&](size_t bytes) -> char* {
        char* p = ws + off;
        off = (off + bytes + 255) & ~(size_t)255;
        return p;
    };
    float*  dis  = (float*)carve((size_t)N * 4);            // deg -> deg^-1/2
    __bf16* w1t  = (__bf16*)carve((size_t)IN * H * 2);      // [H,IN]
    __bf16* w2t  = (__bf16*)carve((size_t)H * H * 2);       // [H,H]
    __bf16* w3t  = (__bf16*)carve((size_t)H * OUT * 2);     // [OUT,H]
    float*  bufA = (float*)carve((size_t)N * H * 4);        // GEMM output
    float*  bufB = (float*)carve((size_t)N * H * 4);        // aggregation output
    __bf16* bufH = (__bf16*)carve((size_t)N * H * 2);       // bf16 activations
    (void)ws_size;

    const int BS = 256;

    // ---- normalization: deg (self-loop + edges) -> deg^-1/2 ----
    k_fill_ones<<<blocks_for(N, BS), BS, 0, stream>>>(dis, N);
    k_deg_accum<<<blocks_for(E, BS), BS, 0, stream>>>(dst, dis, E);
    k_rsqrt_inplace<<<blocks_for(N, BS), BS, 0, stream>>>(dis, N);

    // ---- weights: transpose + bf16 ----
    k_transpose_cvt<<<blocks_for((long long)IN * H, BS), BS, 0, stream>>>(W1, w1t, IN, H);
    k_transpose_cvt<<<blocks_for((long long)H * H,  BS), BS, 0, stream>>>(W2, w2t, H, H);
    k_transpose_cvt<<<blocks_for((long long)H * OUT,BS), BS, 0, stream>>>(W3, w3t, H, OUT);

    // ---- layer 1: h1 = GCN(x, W1, b1); lrelu ----
    k_cvt_bf16<<<blocks_for((long long)N * IN, BS), BS, 0, stream>>>(x, bufH, (long long)N * IN);
    {
        unsigned tiles = (unsigned)((N / 16) * (H / 16));
        gcn_wmma_gemm_bf16<<<(tiles + 7) / 8, BS, 0, stream>>>(bufH, w1t, bufA, N, H, IN);
    }
    k_agg_init<<<blocks_for((long long)N * H, BS), BS, 0, stream>>>(bufA, dis, b1, bufB, N, H);
    k_edge_scatter<<<blocks_for((long long)E * (H / 4), BS), BS, 0, stream>>>(bufA, dis, src, dst, bufB, E, H);
    k_lrelu_cvt_bf16<<<blocks_for((long long)N * H, BS), BS, 0, stream>>>(bufB, bufH, (long long)N * H);

    // ---- layer 2: h2 = GCN(h1, W2, b2); lrelu ----
    {
        unsigned tiles = (unsigned)((N / 16) * (H / 16));
        gcn_wmma_gemm_bf16<<<(tiles + 7) / 8, BS, 0, stream>>>(bufH, w2t, bufA, N, H, H);
    }
    k_agg_init<<<blocks_for((long long)N * H, BS), BS, 0, stream>>>(bufA, dis, b2, bufB, N, H);
    k_edge_scatter<<<blocks_for((long long)E * (H / 4), BS), BS, 0, stream>>>(bufA, dis, src, dst, bufB, E, H);
    k_lrelu_cvt_bf16<<<blocks_for((long long)N * H, BS), BS, 0, stream>>>(bufB, bufH, (long long)N * H);

    // ---- layer 3: h3 = GCN(h2, W3, b3) (no activation) ----
    {
        unsigned tiles = (unsigned)((N / 16) * (OUT / 16));
        gcn_wmma_gemm_bf16<<<(tiles + 7) / 8, BS, 0, stream>>>(bufH, w3t, bufA, N, OUT, H);
    }
    k_agg_init<<<blocks_for((long long)N * OUT, BS), BS, 0, stream>>>(bufA, dis, b3, bufB, N, OUT);
    k_edge_scatter<<<blocks_for((long long)E * (OUT / 4), BS), BS, 0, stream>>>(bufA, dis, src, dst, bufB, E, OUT);

    // ---- readout: first node per graph ----
    k_gather_first<<<blocks_for(N, BS), BS, 0, stream>>>(bufB, batch, out, N, OUT);
    (void)out_size; (void)n_in;
}
